// GCNMD_36335423324414
// MI455X (gfx1250) — matrix operand order repaired and checked by reading, compiled, not run on hardware
//
#include <hip/hip_runtime.h>
#include <hip/hip_bf16.h>
#include <math.h>

// ---------------------------------------------------------------------------
// GCN (two modalities) on MI455X / gfx1250 (wave32, WMMA)
//   out = log_softmax( relu(gcn(x1,W1,b1)) + relu(gcn(x2,W2,b2)) )
// GEMMs via v_wmma_f32_16x16x32_bf16 (fp32 accumulate); scatter via
// global_atomic_add_f32; log-softmax via wave32 shuffle reductions.
// ---------------------------------------------------------------------------

typedef __attribute__((ext_vector_type(16))) __bf16 v16bf;
typedef __attribute__((ext_vector_type(8)))  float  v8f;

#define OUTC 128      // output feature dim (128)
#define TM   128      // M tile per block
#define KT   32       // K tile (one bf16 WMMA K-depth)

// ---------------------------------------------------------------- init ------
__global__ __launch_bounds__(256)
void gcn_init(float* __restrict__ acc1, float* __restrict__ acc2,
              float* __restrict__ deg, int n_nodes)
{
    long long idx = (long long)blockIdx.x * blockDim.x + threadIdx.x;
    long long tot = (long long)n_nodes * OUTC;
    if (idx < tot) { acc1[idx] = 0.0f; acc2[idx] = 0.0f; }
    if (idx < n_nodes) deg[idx] = 1.0f;   // self-loop contribution
}

// ------------------------------------------------------------- degrees ------
__global__ __launch_bounds__(256)
void gcn_degree(const long long* __restrict__ ei, float* __restrict__ deg, int E)
{
    long long e = (long long)blockIdx.x * blockDim.x + threadIdx.x;
    if (e >= E) return;
    int dst = (int)ei[E + e];            // edge_index[1][e]
    atomicAdd(&deg[dst], 1.0f);
}

__global__ __launch_bounds__(256)
void gcn_dinv(float* __restrict__ deg, int n_nodes)
{
    long long i = (long long)blockIdx.x * blockDim.x + threadIdx.x;
    if (i < n_nodes) deg[i] = rsqrtf(deg[i]);   // deg >= 1 always
}

// ------------------------------------------------- WMMA bf16 GEMM -----------
// XW[N x 128] = X[N x K] * W[K x 128]; f32 in/out, bf16 WMMA, fp32 accumulate.
// Block: 256 threads (8 waves). Each block: 128 rows x 128 cols.
// Wave w owns rows w*16..w*16+15 -> 8 accumulator tiles of 16x16.
__global__ __launch_bounds__(256)
void gcn_gemm_wmma(const float* __restrict__ X, const float* __restrict__ W,
                   float* __restrict__ XW, int n_rows, int K)
{
    __shared__ __bf16 lA[TM][KT + 2];        // padded to dodge bank conflicts
    __shared__ __bf16 lB[KT][OUTC + 8];

    const int t    = threadIdx.x;
    const int wave = t >> 5;
    const int lane = t & 31;
    const int half = lane >> 4;              // 0: lanes 0-15, 1: lanes 16-31
    const int l16  = lane & 15;
    const int mBase = blockIdx.x * TM;

    v8f acc[8];
    #pragma unroll
    for (int i = 0; i < 8; ++i) { v8f z = {}; acc[i] = z; }

    for (int k0 = 0; k0 < K; k0 += KT) {
        // --- stage A tile (128 x 32 f32 -> bf16), coalesced over K ---
        {
            int c = t & 31;
            #pragma unroll 4
            for (int r = (t >> 5); r < TM; r += 8) {
                int gr = mBase + r;
                if (gr >= n_rows) gr = n_rows - 1;          // clamp (stores masked)
                lA[r][c] = (__bf16)X[(size_t)gr * K + k0 + c];
            }
        }
        // --- stage B tile (32 x 128 f32 -> bf16), coalesced over cols ---
        {
            int c = t & 127;
            #pragma unroll
            for (int kk = (t >> 7); kk < KT; kk += 2)
                lB[kk][c] = (__bf16)W[(size_t)(k0 + kk) * OUTC + c];
        }
        __syncthreads();

        // --- A fragment: ISA 16-bit A 16x32 layout ---
        // lanes 0-15: K in {0..7, 16..23}; lanes 16-31: K in {8..15, 24..31}
        v16bf a;
        {
            int row = wave * 16 + l16;
            #pragma unroll
            for (int i = 0; i < 8; ++i) {
                a[i]     = lA[row][half * 8 + i];
                a[8 + i] = lA[row][16 + half * 8 + i];
            }
        }

        // --- 8 column tiles, B fragment: N across lanes, K across values (same K order as A) ---
        #pragma unroll
        for (int ct = 0; ct < 8; ++ct) {
            v16bf b;
            #pragma unroll
            for (int j = 0; j < 8; ++j) {
                b[j]     = lB[half * 8 + j][ct * 16 + l16];
                b[8 + j] = lB[16 + half * 8 + j][ct * 16 + l16];
            }
            acc[ct] = __builtin_amdgcn_wmma_f32_16x16x32_bf16(
                false, a, false, b, (short)0, acc[ct], false, false);
        }
        __syncthreads();
    }

    // --- store: C layout = VGPR r -> row (half*8 + r), lane&15 -> col ---
    #pragma unroll
    for (int ct = 0; ct < 8; ++ct) {
        #pragma unroll
        for (int r = 0; r < 8; ++r) {
            int row = mBase + wave * 16 + half * 8 + r;
            int col = ct * 16 + l16;
            if (row < n_rows) XW[(size_t)row * OUTC + col] = acc[ct][r];
        }
    }
}

// ------------------------------------------------------ edge scatter --------
// One wave per edge (incl. N trailing self-loops). Each lane: 4 channels.
__global__ __launch_bounds__(256)
void gcn_scatter(const float* __restrict__ xw1, const float* __restrict__ xw2,
                 const float* __restrict__ dinv, const long long* __restrict__ ei,
                 int E, int n_nodes,
                 float* __restrict__ acc1, float* __restrict__ acc2)
{
    long long slot = (long long)blockIdx.x * 8 + (threadIdx.x >> 5);
    int lane = threadIdx.x & 31;
    long long total = (long long)E + n_nodes;
    if (slot >= total) return;

    int src, dst;
    if (slot < E) { src = (int)ei[slot]; dst = (int)ei[E + slot]; }
    else          { src = dst = (int)(slot - E); }           // self-loop

    float norm = dinv[src] * dinv[dst];

    const float4* r1 = (const float4*)(xw1 + (size_t)src * OUTC);
    const float4* r2 = (const float4*)(xw2 + (size_t)src * OUTC);
    float4 v1 = r1[lane];
    float4 v2 = r2[lane];

    float* o1 = acc1 + (size_t)dst * OUTC + lane * 4;
    float* o2 = acc2 + (size_t)dst * OUTC + lane * 4;
    atomicAdd(o1 + 0, v1.x * norm);
    atomicAdd(o1 + 1, v1.y * norm);
    atomicAdd(o1 + 2, v1.z * norm);
    atomicAdd(o1 + 3, v1.w * norm);
    atomicAdd(o2 + 0, v2.x * norm);
    atomicAdd(o2 + 1, v2.y * norm);
    atomicAdd(o2 + 2, v2.z * norm);
    atomicAdd(o2 + 3, v2.w * norm);
}

// ------------------------------------------- relu+add+log_softmax -----------
// One wave per node row; lane holds 4 channels (float4). acc1 aliases d_out:
// each lane reads its own slot before overwriting -> safe.
__global__ __launch_bounds__(256)
void gcn_epilogue(float* __restrict__ out /* == acc1 */,
                  const float* __restrict__ acc2,
                  const float* __restrict__ b1, const float* __restrict__ b2,
                  int n_nodes)
{
    int wave = threadIdx.x >> 5;
    int lane = threadIdx.x & 31;
    long long row = (long long)blockIdx.x * 8 + wave;
    if (row >= n_nodes) return;

    float4 v1 = ((const float4*)(out  + row * OUTC))[lane];
    float4 v2 = ((const float4*)(acc2 + row * OUTC))[lane];
    float4 c1 = ((const float4*)b1)[lane];
    float4 c2 = ((const float4*)b2)[lane];

    float4 v;
    v.x = fmaxf(v1.x + c1.x, 0.0f) + fmaxf(v2.x + c2.x, 0.0f);
    v.y = fmaxf(v1.y + c1.y, 0.0f) + fmaxf(v2.y + c2.y, 0.0f);
    v.z = fmaxf(v1.z + c1.z, 0.0f) + fmaxf(v2.z + c2.z, 0.0f);
    v.w = fmaxf(v1.w + c1.w, 0.0f) + fmaxf(v2.w + c2.w, 0.0f);

    float m = fmaxf(fmaxf(v.x, v.y), fmaxf(v.z, v.w));
    #pragma unroll
    for (int off = 16; off > 0; off >>= 1)
        m = fmaxf(m, __shfl_xor(m, off, 32));

    float s = __expf(v.x - m) + __expf(v.y - m) + __expf(v.z - m) + __expf(v.w - m);
    #pragma unroll
    for (int off = 16; off > 0; off >>= 1)
        s += __shfl_xor(s, off, 32);

    float lse = m + __logf(s);
    float4 o = { v.x - lse, v.y - lse, v.z - lse, v.w - lse };
    ((float4*)(out + row * OUTC))[lane] = o;
}

// ---------------------------------------------------------------------------
extern "C" void kernel_launch(void* const* d_in, const int* in_sizes, int n_in,
                              void* d_out, int out_size, void* d_ws, size_t ws_size,
                              hipStream_t stream)
{
    const float*      x1 = (const float*)d_in[0];
    const float*      x2 = (const float*)d_in[1];
    const long long*  ei = (const long long*)d_in[2];   // int64 edge_index [2,E]
    const float*      W1 = (const float*)d_in[3];
    const float*      b1 = (const float*)d_in[4];
    const float*      W2 = (const float*)d_in[5];
    const float*      b2 = (const float*)d_in[6];
    float*            out = (float*)d_out;

    const int K1 = in_sizes[3] / OUTC;          // 512
    const int K2 = in_sizes[5] / OUTC;          // 256
    const int N  = in_sizes[0] / K1;            // 50000
    const int E  = in_sizes[2] / 2;             // 800000

    // Workspace layout (floats): deg | xw1 | xw2 | acc2. acc1 aliases d_out.
    float* ws   = (float*)d_ws;
    size_t degN = ((size_t)N + 255) & ~(size_t)255;
    float* deg  = ws;
    float* xw1  = ws + degN;
    float* xw2  = xw1 + (size_t)N * OUTC;
    float* acc2 = xw2 + (size_t)N * OUTC;
    float* acc1 = out;

    // 1) init accumulators + deg
    {
        long long tot = (long long)N * OUTC;
        int blocks = (int)((tot + 255) / 256);
        gcn_init<<<blocks, 256, 0, stream>>>(acc1, acc2, deg, N);
    }
    // 2) degree + dinv
    gcn_degree<<<(E + 255) / 256, 256, 0, stream>>>(ei, deg, E);
    gcn_dinv<<<(N + 255) / 256, 256, 0, stream>>>(deg, N);
    // 3) dense GEMMs (WMMA bf16, fp32 accumulate)
    {
        int mblocks = (N + TM - 1) / TM;
        gcn_gemm_wmma<<<mblocks, 256, 0, stream>>>(x1, W1, xw1, N, K1);
        gcn_gemm_wmma<<<mblocks, 256, 0, stream>>>(x2, W2, xw2, N, K2);
    }
    // 4) normalized scatter-add over edges + self-loops
    {
        long long total = (long long)E + N;
        int blocks = (int)((total + 7) / 8);
        gcn_scatter<<<blocks, 256, 0, stream>>>(xw1, xw2, deg, ei, E, N, acc1, acc2);
    }
    // 5) relu + bias + add + log_softmax
    gcn_epilogue<<<(N + 7) / 8, 256, 0, stream>>>(acc1, acc2, b1, b2, N);
}
